// IMG2SVG_57183194579068
// MI455X (gfx1250) — compile-verified, matrix operand order checked by reading, not run on hardware
//
#include <hip/hip_runtime.h>
#include <hip/hip_bf16.h>

typedef __attribute__((ext_vector_type(16))) _Float16 v16h;
typedef __attribute__((ext_vector_type(8)))  float    v8f;
typedef __attribute__((ext_vector_type(4)))  int      v4i;

typedef __attribute__((address_space(1))) v4i gv4i;   // global int4
typedef __attribute__((address_space(3))) v4i lv4i;   // LDS int4

union FragU { v16h h; uint4 q[2]; };

#define NIMG 16384

// ---- CDNA5 async global->LDS path (guarded; falls back to plain copy) ----
#if defined(__has_builtin)
#  if __has_builtin(__builtin_amdgcn_global_load_async_to_lds_b128)
#    define USE_ASYNC_LDS 1
#  endif
#endif

#if defined(__has_builtin)
#  if __has_builtin(__builtin_amdgcn_s_wait_asynccnt)
#    define WAIT_ASYNC() __builtin_amdgcn_s_wait_asynccnt(0)
#  else
#    define WAIT_ASYNC() asm volatile("s_wait_asynccnt 0x0" ::: "memory")
#  endif
#else
#  define WAIT_ASYNC() asm volatile("s_wait_asynccnt 0x0" ::: "memory")
#endif

__device__ inline v8f wmma_f16(v16h a, v16h b, v8f c) {
    return __builtin_amdgcn_wmma_f32_16x16x32_f16(false, a, false, b, (short)0, c, false, false);
}

// A-fragment from LDS: off (halves) = lane-dependent base + constant tap/chunk
// offset. Lane group 0 needs K=0..7,16..23; group 1 needs K=8..15,24..31 ->
// two 16B ds_load_b128 chunks 32B apart (constant imm offsets after unroll).
__device__ inline v16h loadAfragOff(const _Float16* lds, int off) {
    FragU f;
    const uint4* p = (const uint4*)(lds + off);
    f.q[0] = p[0];
    f.q[1] = p[2];   // +16 halves
    return f.h;
}

// Packed B fragments: frag index (s*NT+nt), 32 lanes x 32 contiguous bytes.
__device__ inline v16h loadBfrag(const uint4* packed, int s, int nt, int NT, int lane) {
    FragU f;
    const uint4* p = packed + ((size_t)((s * NT + nt) * 32 + lane)) * 2;
    f.q[0] = p[0];
    f.q[1] = p[1];
    return f.h;
}

// ---------------------------------------------------------------------------
// Weight packer: HWIO fp32 -> f16 WMMA B-fragment layout.
// K = (ky*3+kx)*Cin + ci ; N = co. One thread per f16 element.
// ---------------------------------------------------------------------------
__global__ void packW_kernel(const float* __restrict__ w, _Float16* __restrict__ out,
                             int Cin, int Cout, int KT, int NT) {
    int idx = blockIdx.x * blockDim.x + threadIdx.x;
    int total = KT * NT * 32 * 16;
    if (idx >= total) return;
    int i    = idx & 15;
    int lane = (idx >> 4) & 31;
    int nt   = (idx >> 9) % NT;
    int kt   = (idx >> 9) / NT;
    int grp  = lane >> 4;
    int kl   = (i < 8) ? (grp * 8 + i) : (16 + grp * 8 + (i - 8));
    int kG   = kt * 32 + kl;
    int tap  = kG / Cin;
    int ci   = kG % Cin;
    int co   = nt * 16 + (lane & 15);
    out[idx] = (_Float16)w[(tap * Cin + ci) * Cout + co];
}

// ---------------------------------------------------------------------------
// Stage A: conv1 (scalar, K=9) -> conv2 (WMMA, K=1152, N=128) -> 2x2 maxpool.
// One block per image; 8 waves, each wave owns one 16-wide N-tile of conv2.
// LDS: 12x12x128 haloed act1 (36.8KB) + 10x10x128 act2 (25.6KB) = 62.5KB.
// ---------------------------------------------------------------------------
__global__ void __launch_bounds__(256) stageA_kernel(
        const float* __restrict__ in, const float* __restrict__ w1,
        const float* __restrict__ b1, const float* __restrict__ b2,
        const uint4* __restrict__ w2p, _Float16* __restrict__ p1out) {
    __shared__ _Float16 act1[12 * 12 * 128];
    __shared__ _Float16 act2[100 * 128];
    const int img = blockIdx.x;
    const int tid = threadIdx.x;

    // zero halo buffer
    for (int i = tid; i < 12 * 12 * 128; i += 256) act1[i] = (_Float16)0.f;
    __syncthreads();

    // conv1: 100 pixels x 128 ch, 9 MACs each
    const float* src = in + (size_t)img * 100;
    for (int i = tid; i < 100 * 128; i += 256) {
        int co = i & 127, p = i >> 7;
        int y = p / 10, x = p % 10;
        float acc = b1[co];
#pragma unroll
        for (int ky = 0; ky < 3; ky++)
#pragma unroll
            for (int kx = 0; kx < 3; kx++) {
                int iy = y + ky - 1, ix = x + kx - 1;
                if (iy >= 0 && iy < 10 && ix >= 0 && ix < 10)
                    acc += src[iy * 10 + ix] * w1[(ky * 3 + kx) * 128 + co];
            }
        act1[((y + 1) * 12 + (x + 1)) * 128 + co] = (_Float16)fmaxf(acc, 0.f);
    }
    __syncthreads();

    // conv2 implicit GEMM: M=100(->7x16), N=128(8 waves x 16), K=1152(36x32)
    const int lane = tid & 31, nt = tid >> 5;
    const int m = lane & 15, grp = lane >> 4;
    const int n = nt * 16 + m;
    const float bias = b2[n];

    // per-mt lane-dependent LDS base (halves); tap/chunk part is a constant
    int aoff[7];
#pragma unroll
    for (int mt = 0; mt < 7; mt++) {
        int p = mt * 16 + m; if (p > 99) p = 99;   // pad rows read pixel 99 (unused)
        int y = p / 10, x = p % 10;
        aoff[mt] = (y * 12 + x) * 128 + grp * 8;
    }
    // bias folded into accumulator init (D row-constant per lane's N)
    v8f acc[7];
#pragma unroll
    for (int mt = 0; mt < 7; mt++)
        acc[mt] = (v8f){bias, bias, bias, bias, bias, bias, bias, bias};

#pragma unroll
    for (int s = 0; s < 36; s++) {
        const int tap = s >> 2, cb = (s & 3) << 5;
        const int ky = tap / 3, kx = tap % 3;
        const int koff = (ky * 12 + kx) * 128 + cb;    // compile-time constant
        v16h b = loadBfrag(w2p, s, nt, 8, lane);
        if (s + 1 < 36)  // near-scope (WGP) prefetch: weights reused by all blocks
            __builtin_prefetch(w2p + (size_t)(((s + 1) * 8 + nt) * 32) * 2, 0, 3);
#pragma unroll
        for (int mt = 0; mt < 7; mt++) {
            v16h a = loadAfragOff(act1, aoff[mt] + koff);
            acc[mt] = wmma_f16(a, b, acc[mt]);
        }
    }
    // D layout: lane L, VGPR r -> M = r + 8*(L>=16), N = L%16 (+16*nt)
#pragma unroll
    for (int mt = 0; mt < 7; mt++) {
#pragma unroll
        for (int r = 0; r < 8; r++) {
            int p = mt * 16 + r + (grp << 3);
            if (p < 100)
                act2[p * 128 + n] = (_Float16)fmaxf(acc[mt][r], 0.f);
        }
    }
    __syncthreads();

    // 2x2 maxpool -> (5,5,128) f16 to HBM
    for (int i = tid; i < 5 * 5 * 128; i += 256) {
        int c = i & 127, pp = i >> 7;
        int oy = pp / 5, ox = pp % 5;
        float v = (float)act2[((2 * oy) * 10 + 2 * ox) * 128 + c];
        v = fmaxf(v, (float)act2[((2 * oy) * 10 + 2 * ox + 1) * 128 + c]);
        v = fmaxf(v, (float)act2[((2 * oy + 1) * 10 + 2 * ox) * 128 + c]);
        v = fmaxf(v, (float)act2[((2 * oy + 1) * 10 + 2 * ox + 1) * 128 + c]);
        p1out[(size_t)img * 3200 + i] = (_Float16)v;
    }
}

// ---------------------------------------------------------------------------
// Stage B: conv3 (K=1152,N=64) -> conv4 (K=576,N=64) -> 2x2 pool -> GAP.
// Wave-per-image (2 images / 64-thread block). LDS 22KB per image.
// Input staged with GLOBAL_LOAD_ASYNC_TO_LDS_B128 (ASYNCcnt path).
// ---------------------------------------------------------------------------
__global__ void __launch_bounds__(64) stageB_kernel(
        const _Float16* __restrict__ p1, const uint4* __restrict__ w3p,
        const uint4* __restrict__ w4p, const float* __restrict__ b3,
        const float* __restrict__ b4, float* __restrict__ gap) {
    __shared__ _Float16 lds[2 * 11008];
    const int wave = threadIdx.x >> 5, lane = threadIdx.x & 31;
    const int img = blockIdx.x * 2 + wave;
    _Float16* inpad = lds + wave * 11008;     // 7x7x128 haloed
    _Float16* a3    = inpad + 6272;           // 7x7x64 haloed
    _Float16* a4    = a3 + 3136;              // 5x5x64
    const int m = lane & 15, grp = lane >> 4;

    // Zero ONLY the halo of inpad (so DS-counter stores and ASYNC-counter
    // writes never alias: the two counters are unordered w.r.t. each other).
    for (int i = lane; i < 6272; i += 32) {
        int pix = i >> 7;
        int y = pix / 7, x = pix % 7;
        if (y == 0 || y == 6 || x == 0 || x == 6) inpad[i] = (_Float16)0.f;
    }
    for (int i = lane; i < 3136; i += 32) a3[i] = (_Float16)0.f;

    // Stage (5,5,128) f16 interior into haloed LDS tile.
    const _Float16* src = p1 + (size_t)img * 3200;
#if defined(USE_ASYNC_LDS)
    for (int i = lane; i < 400; i += 32) {           // 400 x 16B chunks
        int pix = i >> 4, c = (i & 15) << 3;
        int y = pix / 5, x = pix % 5;
        __builtin_amdgcn_global_load_async_to_lds_b128(
            (gv4i*)(src + (size_t)i * 8),
            (lv4i*)&inpad[((y + 1) * 7 + (x + 1)) * 128 + c],
            0, 0);
    }
    WAIT_ASYNC();
#else
    for (int i = lane; i < 3200; i += 32) {
        int c = i & 127, pix = i >> 7;
        int y = pix / 5, x = pix % 5;
        inpad[((y + 1) * 7 + (x + 1)) * 128 + c] = src[i];
    }
#endif

    // lane-dependent A bases (halves) for the two M-tiles
    const int p0 = m;
    const int p1i = (16 + m > 24) ? 24 : 16 + m;
    const int y0 = p0 / 5, x0 = p0 % 5;
    const int y1 = p1i / 5, x1 = p1i % 5;
    const int a3off0 = (y0 * 7 + x0) * 128 + grp * 8;
    const int a3off1 = (y1 * 7 + x1) * 128 + grp * 8;
    const int a4off0 = (y0 * 7 + x0) * 64 + grp * 8;
    const int a4off1 = (y1 * 7 + x1) * 64 + grp * 8;

    // conv3: M=25(->2x16), N=64(4 tiles), K=1152(36x32)
    for (int nt = 0; nt < 4; nt++) {
        int n = nt * 16 + m;
        float bias = b3[n];
        v8f acc0 = (v8f){bias, bias, bias, bias, bias, bias, bias, bias};
        v8f acc1 = acc0;
#pragma unroll
        for (int s = 0; s < 36; s++) {
            const int tap = s >> 2, cb = (s & 3) << 5;
            const int ky = tap / 3, kx = tap % 3;
            const int koff = (ky * 7 + kx) * 128 + cb;   // constant
            v16h b = loadBfrag(w3p, s, nt, 4, lane);
            if (s + 1 < 36)
                __builtin_prefetch(w3p + (size_t)(((s + 1) * 4 + nt) * 32) * 2, 0, 3);
            v16h a0 = loadAfragOff(inpad, a3off0 + koff);
            acc0 = wmma_f16(a0, b, acc0);
            v16h a1 = loadAfragOff(inpad, a3off1 + koff);
            acc1 = wmma_f16(a1, b, acc1);
        }
#pragma unroll
        for (int r = 0; r < 8; r++) {
            int M = r + (grp << 3);
            if (M < 25) {
                int y = M / 5, x = M % 5;
                a3[((y + 1) * 7 + (x + 1)) * 64 + n] = (_Float16)fmaxf(acc0[r], 0.f);
            }
            int p = 16 + M;
            if (p < 25) {
                int y = p / 5, x = p % 5;
                a3[((y + 1) * 7 + (x + 1)) * 64 + n] = (_Float16)fmaxf(acc1[r], 0.f);
            }
        }
    }

    // conv4: M=25, N=64, K=576(18x32)
    for (int nt = 0; nt < 4; nt++) {
        int n = nt * 16 + m;
        float bias = b4[n];
        v8f acc0 = (v8f){bias, bias, bias, bias, bias, bias, bias, bias};
        v8f acc1 = acc0;
#pragma unroll
        for (int s = 0; s < 18; s++) {
            const int tap = s >> 1, cb = (s & 1) << 5;
            const int ky = tap / 3, kx = tap % 3;
            const int koff = (ky * 7 + kx) * 64 + cb;    // constant
            v16h b = loadBfrag(w4p, s, nt, 4, lane);
            if (s + 1 < 18)
                __builtin_prefetch(w4p + (size_t)(((s + 1) * 4 + nt) * 32) * 2, 0, 3);
            v16h a0 = loadAfragOff(a3, a4off0 + koff);
            acc0 = wmma_f16(a0, b, acc0);
            v16h a1 = loadAfragOff(a3, a4off1 + koff);
            acc1 = wmma_f16(a1, b, acc1);
        }
#pragma unroll
        for (int r = 0; r < 8; r++) {
            int M = r + (grp << 3);
            if (M < 25) a4[M * 64 + n] = (_Float16)fmaxf(acc0[r], 0.f);
            int p = 16 + M;
            if (p < 25) a4[p * 64 + n] = (_Float16)fmaxf(acc1[r], 0.f);
        }
    }

    // 2x2 pool (VALID over 5x5 -> 2x2) then mean -> (64,) f32
    for (int c = lane; c < 64; c += 32) {
        float sum = 0.f;
#pragma unroll
        for (int oy = 0; oy < 2; oy++)
#pragma unroll
            for (int ox = 0; ox < 2; ox++) {
                float v = (float)a4[((2 * oy) * 5 + 2 * ox) * 64 + c];
                v = fmaxf(v, (float)a4[((2 * oy) * 5 + 2 * ox + 1) * 64 + c]);
                v = fmaxf(v, (float)a4[((2 * oy + 1) * 5 + 2 * ox) * 64 + c]);
                v = fmaxf(v, (float)a4[((2 * oy + 1) * 5 + 2 * ox + 1) * 64 + c]);
                sum += v;
            }
        gap[(size_t)img * 64 + c] = sum * 0.25f;
    }
}

// ---------------------------------------------------------------------------
// Stage C: dense 64->32 relu, 32->4 sigmoid, polyline round/sort/compact.
// One thread per image.
// ---------------------------------------------------------------------------
__global__ void __launch_bounds__(256) stageC_kernel(
        const float* __restrict__ gap, const float* __restrict__ wd1,
        const float* __restrict__ bd1, const float* __restrict__ wd2,
        const float* __restrict__ bd2, float* __restrict__ out) {
    int img = blockIdx.x * 256 + threadIdx.x;
    if (img >= NIMG) return;
    const float* g = gap + (size_t)img * 64;

    float h[32];
#pragma unroll 4
    for (int o = 0; o < 32; o++) {
        float a = bd1[o];
        for (int i = 0; i < 64; i++) a += g[i] * wd1[i * 32 + o];
        h[o] = fmaxf(a, 0.f);
    }
    float s4[4];
#pragma unroll
    for (int o = 0; o < 4; o++) {
        float a = bd2[o];
        for (int i = 0; i < 32; i++) a += h[i] * wd2[i * 4 + o];
        s4[o] = 1.f / (1.f + __expf(-a));
    }
    // pts: interpolate p1=(s0,s1) -> p2=(s2,s3), scale by 10, RNE round
    float px[5], py[5], pos[5];
#pragma unroll
    for (int k = 0; k < 5; k++) {
        float t = 0.25f * (float)k;
        float X = rintf(((1.f - t) * s4[0] + t * s4[2]) * 10.f);
        float Y = rintf(((1.f - t) * s4[1] + t * s4[3]) * 10.f);
        px[k] = X; py[k] = Y; pos[k] = X * 10.f + Y;
    }
    // stable argsort by pos (insertion sort)
    int ord[5] = {0, 1, 2, 3, 4};
    for (int i = 1; i < 5; i++) {
        int k = ord[i];
        float pk = pos[k];
        int j = i - 1;
        while (j >= 0 && pos[ord[j]] > pk) { ord[j + 1] = ord[j]; j--; }
        ord[j + 1] = k;
    }
    // dedup-compact: keep k==0 or L1 diff != 0
    float* op = out + (size_t)img * 10;
#pragma unroll
    for (int i = 0; i < 10; i++) op[i] = -1.f;
    int idx = 0;
    float lx = 0.f, ly = 0.f;
    for (int k = 0; k < 5; k++) {
        float sx = px[ord[k]], sy = py[ord[k]];
        float d = (k == 0) ? 1.f : (fabsf(sx - lx) + fabsf(sy - ly));
        lx = sx; ly = sy;
        if (d != 0.f) { op[idx * 2] = sx; op[idx * 2 + 1] = sy; idx++; }
    }
}

// ---------------------------------------------------------------------------
extern "C" void kernel_launch(void* const* d_in, const int* in_sizes, int n_in,
                              void* d_out, int out_size, void* d_ws, size_t ws_size,
                              hipStream_t stream) {
    const float* data = (const float*)d_in[0];
    const float* w1   = (const float*)d_in[1];
    const float* b1   = (const float*)d_in[2];
    const float* w2   = (const float*)d_in[3];
    const float* b2   = (const float*)d_in[4];
    const float* w3   = (const float*)d_in[5];
    const float* b3   = (const float*)d_in[6];
    const float* w4   = (const float*)d_in[7];
    const float* b4   = (const float*)d_in[8];
    const float* wd1  = (const float*)d_in[9];
    const float* bd1  = (const float*)d_in[10];
    const float* wd2  = (const float*)d_in[11];
    const float* bd2  = (const float*)d_in[12];
    (void)in_sizes; (void)n_in; (void)out_size; (void)ws_size;

    // Workspace layout (bytes):
    //   [0)        w2 packed f16 : 36*8*512*2  = 294912
    //   [294912)   w3 packed f16 : 36*4*512*2  = 147456
    //   [442368)   w4 packed f16 : 18*4*512*2  =  73728
    //   [516096)   p1 (B,5,5,128) f16          = 104857600
    //   [+)        gap (B,64) f32              = 4194304    (~110 MB total)
    char* ws = (char*)d_ws;
    _Float16* w2p = (_Float16*)(ws);
    _Float16* w3p = (_Float16*)(ws + 294912);
    _Float16* w4p = (_Float16*)(ws + 442368);
    _Float16* p1b = (_Float16*)(ws + 516096);
    float*    gapb = (float*)(ws + 516096 + 104857600ULL);

    packW_kernel<<<(36 * 8 * 512 + 255) / 256, 256, 0, stream>>>(w2, w2p, 128, 128, 36, 8);
    packW_kernel<<<(36 * 4 * 512 + 255) / 256, 256, 0, stream>>>(w3, w3p, 128, 64, 36, 4);
    packW_kernel<<<(18 * 4 * 512 + 255) / 256, 256, 0, stream>>>(w4, w4p, 64, 64, 18, 4);

    stageA_kernel<<<NIMG, 256, 0, stream>>>(data, w1, b1, b2, (const uint4*)w2p, p1b);
    stageB_kernel<<<NIMG / 2, 64, 0, stream>>>(p1b, (const uint4*)w3p, (const uint4*)w4p, b3, b4, gapb);
    stageC_kernel<<<(NIMG + 255) / 256, 256, 0, stream>>>(gapb, wd1, bd1, wd2, bd2, (float*)d_out);
}
